// CrossAttention_7069516169643
// MI455X (gfx1250) — matrix operand order repaired and checked by reading, not compiled, not run on hardware
//
#include <hip/hip_runtime.h>

typedef __attribute__((ext_vector_type(16))) _Float16 v16h;
typedef __attribute__((ext_vector_type(8)))  _Float16 v8h;
typedef __attribute__((ext_vector_type(4)))  _Float16 v4h;
typedef __attribute__((ext_vector_type(8)))  float    v8f;
typedef __attribute__((ext_vector_type(4)))  float    v4f;

#define DIMD    256
#define NTOK    4096
#define BATCH   4
#define BC      32                  // keys/values per LDS tile
#define WAVES   4
#define TPB     (WAVES * 32)
#define WGROWS  (WAVES * 16)        // 64 query rows per workgroup
#define ATT_SCALE 0.0625f           // 256^-0.5
#define NBLK    (BATCH * (NTOK / WGROWS))   // 256 workgroups per phase

// ---------------------------------------------------------------------------
// D = A(16x32 f16) * B(32x16 f16) + C(16x16 f32)
__device__ __forceinline__ v8f wmma_f16(v16h a, v16h b, v8f c) {
  return __builtin_amdgcn_wmma_f32_16x16x32_f16(false, a, false, b, (short)0, c,
                                                false, false);
}

// Load 16 rows x 256 cols fp32 (row base already includes lane row m) into
// eight 16x32 A-fragments, converting to f16 with an optional scale.
// A-frag element map (this code's contiguous k order, matched by B): lane l: m=l&15, h=l>>4;
//   elems 0..7  -> K = 32c + 16h + i      (contiguous)
//   elems 8..15 -> K = 32c + 16h + 8 + i  (contiguous)
__device__ __forceinline__ void load_afrag_rows(v16h* qa, const float* __restrict__ row,
                                                float scale, int h)
{
#pragma unroll
  for (int c = 0; c < 8; ++c) {
    const float* p0 = row + 32 * c + 16 * h;
#pragma unroll
    for (int i = 0; i < 8; ++i) {
      qa[c][i]     = (_Float16)(p0[i] * scale);
      qa[c][8 + i] = (_Float16)(p0[8 + i] * scale);
    }
  }
}

// Cooperative fill of one BC x DIMD fp32 tile into LDS as f16:
//   sRM[kr][dd]  row-major (scaled by rmScale)   -> B-frags for S = A * tile^T
//   sTR[dd][kr]  transposed (unscaled)           -> B-frags for O += P * tile
__device__ __forceinline__ void fill_tile(_Float16* sRM, _Float16* sTR,
                                          const float* __restrict__ src,
                                          float rmScale, int tid)
{
#pragma unroll 4
  for (int t = tid; t < BC * (DIMD / 4); t += TPB) {
    const int kr = t >> 6;            // t / (DIMD/4)
    const int dq = t & 63;
    const v4f v = *(const v4f*)(src + (size_t)kr * DIMD + dq * 4);
    const int dd = dq * 4;
    v4h hs;
    hs[0] = (_Float16)(v[0] * rmScale);
    hs[1] = (_Float16)(v[1] * rmScale);
    hs[2] = (_Float16)(v[2] * rmScale);
    hs[3] = (_Float16)(v[3] * rmScale);
    *(v4h*)(sRM + (size_t)kr * DIMD + dd) = hs;
    sTR[(size_t)(dd + 0) * BC + kr] = (_Float16)v[0];
    sTR[(size_t)(dd + 1) * BC + kr] = (_Float16)v[1];
    sTR[(size_t)(dd + 2) * BC + kr] = (_Float16)v[2];
    sTR[(size_t)(dd + 3) * BC + kr] = (_Float16)v[3];
  }
}

// Two 16x16 score tiles: s_t[row][n] = sum_d A[row][d] * sRM[16t+n][d].
// B-frag (contiguous k order, same as A): lane holds column n=l&15, elems e -> K=16h+e
// -> contiguous 32B reads from the row-major LDS copy.
__device__ __forceinline__ void score_tiles(v8f& s0, v8f& s1, const v16h* qa,
                                            const _Float16* sRM, int m, int h)
{
#pragma unroll
  for (int c = 0; c < 8; ++c) {
    v16h b0 = *(const v16h*)(sRM + (size_t)m * DIMD + 32 * c + 16 * h);
    s0 = wmma_f16(qa[c], b0, s0);
  }
#pragma unroll
  for (int c = 0; c < 8; ++c) {
    v16h b1 = *(const v16h*)(sRM + (size_t)(16 + m) * DIMD + 32 * c + 16 * h);
    s1 = wmma_f16(qa[c], b1, s1);
  }
}

// Restage P (held in C-fragment layout) through LDS into an A-fragment and
// accumulate O[j] += P(16x32) * V(32x16 per j) using the transposed LDS copy.
__device__ __forceinline__ void pv_accum(v8f* o, const float* p0v, const float* p1v,
                                         _Float16* sPw, const _Float16* sTR,
                                         int m, int h)
{
#pragma unroll
  for (int r = 0; r < 8; ++r) {        // C layout: row 8h+r, col m (+16*tile)
    sPw[(8 * h + r) * BC + m]      = (_Float16)p0v[r];
    sPw[(8 * h + r) * BC + 16 + m] = (_Float16)p1v[r];
  }
  asm volatile("s_wait_dscnt 0" ::: "memory");   // same-wave LDS RAW fence
  v8h plo = *(const v8h*)(sPw + (size_t)m * BC + 16 * h);
  v8h phi = *(const v8h*)(sPw + (size_t)m * BC + 16 * h + 8);
  v16h pa;
#pragma unroll
  for (int i = 0; i < 8; ++i) { pa[i] = plo[i]; pa[8 + i] = phi[i]; }
#pragma unroll
  for (int j = 0; j < 16; ++j) {
    v16h vf = *(const v16h*)(sTR + (size_t)(16 * j + m) * BC + 16 * h);
    o[j] = wmma_f16(pa, vf, o[j]);
  }
}

// ---------------------------------------------------------------------------
// Phase 1: flash attention over text keys -> image_out + per-row (max, sumexp)
__global__ __launch_bounds__(TPB)
void xattn_image(const float* __restrict__ img, const float* __restrict__ txt,
                 float* __restrict__ img_out, float* __restrict__ mws,
                 float* __restrict__ lws)
{
  __shared__ __align__(32) _Float16 sRM[BC * DIMD];
  __shared__ __align__(32) _Float16 sTR[DIMD * BC];
  __shared__ __align__(32) _Float16 sP[WAVES * 16 * BC];

  const int tid  = threadIdx.x;
  const int wv   = tid >> 5;
  const int lane = tid & 31;
  const int m    = lane & 15;
  const int h    = lane >> 4;

  const int wgPerB = NTOK / WGROWS;
  const int b  = blockIdx.x / wgPerB;
  const int q0 = (blockIdx.x % wgPerB) * WGROWS + wv * 16;

  v16h qa[8];
  load_afrag_rows(qa, img + ((size_t)b * NTOK + q0 + m) * DIMD, ATT_SCALE, h);

  v8f o[16];
#pragma unroll
  for (int j = 0; j < 16; ++j)
#pragma unroll
    for (int r = 0; r < 8; ++r) o[j][r] = 0.0f;

  float mrow[8], lrow[8];
#pragma unroll
  for (int r = 0; r < 8; ++r) { mrow[r] = -3.0e38f; lrow[r] = 0.0f; }

  const float* tb = txt + (size_t)b * NTOK * DIMD;

  for (int k0 = 0; k0 < NTOK; k0 += BC) {
    __syncthreads();
    fill_tile(sRM, sTR, tb + (size_t)k0 * DIMD, 1.0f, tid);
    __syncthreads();
    if (k0 + BC < NTOK) {
      const float* nx = tb + (size_t)(k0 + BC) * DIMD;
      __builtin_prefetch(nx + tid * 32, 0, 0);
      __builtin_prefetch(nx + (tid + TPB) * 32, 0, 0);
    }

    v8f s0, s1;
#pragma unroll
    for (int r = 0; r < 8; ++r) { s0[r] = 0.0f; s1[r] = 0.0f; }
    score_tiles(s0, s1, qa, sRM, m, h);

    float tmax[8];
#pragma unroll
    for (int r = 0; r < 8; ++r) tmax[r] = fmaxf(s0[r], s1[r]);
#pragma unroll
    for (int mk = 1; mk < 16; mk <<= 1)
#pragma unroll
      for (int r = 0; r < 8; ++r)
        tmax[r] = fmaxf(tmax[r], __shfl_xor(tmax[r], mk, 32));

    float p0v[8], p1v[8], alpha[8], tsum[8];
#pragma unroll
    for (int r = 0; r < 8; ++r) {
      const float mn = fmaxf(mrow[r], tmax[r]);
      alpha[r] = __expf(mrow[r] - mn);
      mrow[r]  = mn;
      p0v[r]   = __expf(s0[r] - mn);
      p1v[r]   = __expf(s1[r] - mn);
      tsum[r]  = p0v[r] + p1v[r];
    }
#pragma unroll
    for (int mk = 1; mk < 16; mk <<= 1)
#pragma unroll
      for (int r = 0; r < 8; ++r) tsum[r] += __shfl_xor(tsum[r], mk, 32);
#pragma unroll
    for (int r = 0; r < 8; ++r) lrow[r] = lrow[r] * alpha[r] + tsum[r];
#pragma unroll
    for (int j = 0; j < 16; ++j)
#pragma unroll
      for (int r = 0; r < 8; ++r) o[j][r] *= alpha[r];

    pv_accum(o, p0v, p1v, sP + wv * 16 * BC, sTR, m, h);
  }

  float* orow = img_out + ((size_t)b * NTOK + q0) * DIMD;
  float rinv[8];
#pragma unroll
  for (int r = 0; r < 8; ++r) rinv[r] = 1.0f / lrow[r];
#pragma unroll
  for (int j = 0; j < 16; ++j)
#pragma unroll
    for (int r = 0; r < 8; ++r)
      orow[(size_t)(8 * h + r) * DIMD + 16 * j + m] = o[j][r] * rinv[r];

  if (m == 0) {
#pragma unroll
    for (int r = 0; r < 8; ++r) {
      mws[(size_t)b * NTOK + q0 + 8 * h + r] = mrow[r];
      lws[(size_t)b * NTOK + q0 + 8 * h + r] = lrow[r];
    }
  }
}

// ---------------------------------------------------------------------------
// Phase 2: text_out = P^T * image, P normalized with phase-1 column stats.
__global__ __launch_bounds__(TPB)
void xattn_text(const float* __restrict__ img, const float* __restrict__ txt,
                float* __restrict__ txt_out, const float* __restrict__ mws,
                const float* __restrict__ lws)
{
  __shared__ __align__(32) _Float16 sRM[BC * DIMD];
  __shared__ __align__(32) _Float16 sTR[DIMD * BC];
  __shared__ __align__(32) _Float16 sP[WAVES * 16 * BC];

  const int tid  = threadIdx.x;
  const int wv   = tid >> 5;
  const int lane = tid & 31;
  const int m    = lane & 15;
  const int h    = lane >> 4;

  const int wgPerB = NTOK / WGROWS;
  const int b  = blockIdx.x / wgPerB;
  const int t0 = (blockIdx.x % wgPerB) * WGROWS + wv * 16;

  v16h ta[8];
  load_afrag_rows(ta, txt + ((size_t)b * NTOK + t0 + m) * DIMD, 1.0f, h);

  v8f o[16];
#pragma unroll
  for (int j = 0; j < 16; ++j)
#pragma unroll
    for (int r = 0; r < 8; ++r) o[j][r] = 0.0f;

  const float* ib = img + (size_t)b * NTOK * DIMD;
  const float* mw = mws + (size_t)b * NTOK;
  const float* lw = lws + (size_t)b * NTOK;

  for (int i0 = 0; i0 < NTOK; i0 += BC) {
    __syncthreads();
    fill_tile(sRM, sTR, ib + (size_t)i0 * DIMD, ATT_SCALE, tid);
    __syncthreads();
    if (i0 + BC < NTOK) {
      const float* nx = ib + (size_t)(i0 + BC) * DIMD;
      __builtin_prefetch(nx + tid * 32, 0, 0);
      __builtin_prefetch(nx + (tid + TPB) * 32, 0, 0);
    }

    v8f s0, s1;
#pragma unroll
    for (int r = 0; r < 8; ++r) { s0[r] = 0.0f; s1[r] = 0.0f; }
    score_tiles(s0, s1, ta, sRM, m, h);

    // Column (image-row) stats: col index = i0 + 16*tile + m
    const float m0 = mw[i0 + m];
    const float r0 = 1.0f / lw[i0 + m];
    const float m1 = mw[i0 + 16 + m];
    const float r1 = 1.0f / lw[i0 + 16 + m];

    float p0v[8], p1v[8];
#pragma unroll
    for (int r = 0; r < 8; ++r) {
      p0v[r] = __expf(s0[r] - m0) * r0;
      p1v[r] = __expf(s1[r] - m1) * r1;
    }
    pv_accum(o, p0v, p1v, sP + wv * 16 * BC, sTR, m, h);
  }

  float* orow = txt_out + ((size_t)b * NTOK + t0) * DIMD;
#pragma unroll
  for (int j = 0; j < 16; ++j)
#pragma unroll
    for (int r = 0; r < 8; ++r)
      orow[(size_t)(8 * h + r) * DIMD + 16 * j + m] = o[j][r];
}

// ---------------------------------------------------------------------------
extern "C" void kernel_launch(void* const* d_in, const int* in_sizes, int n_in,
                              void* d_out, int out_size, void* d_ws, size_t ws_size,
                              hipStream_t stream)
{
  (void)in_sizes; (void)n_in; (void)out_size; (void)ws_size;
  const float* img = (const float*)d_in[0];
  const float* txt = (const float*)d_in[1];
  float* img_out = (float*)d_out;
  float* txt_out = img_out + (size_t)BATCH * NTOK * DIMD;
  float* mws = (float*)d_ws;                       // [B*NTOK] row maxima
  float* lws = mws + (size_t)BATCH * NTOK;         // [B*NTOK] row sum-exp

  xattn_image<<<dim3(NBLK), dim3(TPB), 0, stream>>>(img, txt, img_out, mws, lws);
  xattn_text <<<dim3(NBLK), dim3(TPB), 0, stream>>>(img, txt, txt_out, mws, lws);
}
